// BiESN2D_90228672954700
// MI455X (gfx1250) — compile-verified, matrix operand order checked by reading, not run on hardware
//
#include <hip/hip_runtime.h>
#include <hip/hip_bf16.h>

typedef __attribute__((ext_vector_type(16))) __bf16 v16bf;
typedef __attribute__((ext_vector_type(8)))  __bf16 v8bf;
typedef __attribute__((ext_vector_type(8)))  float  v8f;

#define NT_SEQ 16
#define T_LEN  128
#define C_IN   64
#define U_DIM  256
#define XROW   72    // 64 + 8 pad (keeps 16B alignment, breaks bank collisions)
#define HROW   264   // 256 + 8 pad

// Hardware tanh on gfx1250 (V_TANH_F32, single TRANS32 op); branch-free
// exp-based fallback (also pure trans ops, no EXEC manipulation).
__device__ __forceinline__ float fast_tanh(float a) {
#if __has_builtin(__builtin_amdgcn_tanhf)
    return __builtin_amdgcn_tanhf(a);
#else
    const float e = __expf(2.0f * a);
    return (e - 1.0f) * __builtin_amdgcn_rcpf(e + 1.0f);
#endif
}

// Load a 16x32 bf16 A/B fragment (per CDNA5 16-bit A-matrix layout) from an
// LDS row.  koff already includes the per-lane kb (= 0 or 8) offset.
// dwords 0-3 <- K=[koff, koff+8), dwords 4-7 <- K=[koff+16, koff+24).
__device__ __forceinline__ v16bf load_frag16(const __bf16* row, int koff) {
    v8bf lo = *(const v8bf*)(row + koff);
    v8bf hi = *(const v8bf*)(row + koff + 16);
    return __builtin_shufflevector(lo, hi, 0,1,2,3,4,5,6,7,8,9,10,11,12,13,14,15);
}

__global__ __launch_bounds__(256) void biesn2d_kernel(
    const float* __restrict__ x,
    const float* __restrict__ wk_hf, const float* __restrict__ wr_hf,
    const float* __restrict__ wk_hb, const float* __restrict__ wr_hb,
    const float* __restrict__ wk_vf, const float* __restrict__ wr_vf,
    const float* __restrict__ wk_vb, const float* __restrict__ wr_vb,
    float* __restrict__ out)
{
    __shared__ __attribute__((aligned(16))) __bf16 xlds[2][NT_SEQ][XROW];
    __shared__ __attribute__((aligned(16))) __bf16 hlds[2][NT_SEQ][HROW];

    const int wg    = blockIdx.x;
    const int dir   = wg >> 6;         // 0:h_fwd 1:h_bwd 2:v_fwd 3:v_bwd
    const int stile = wg & 63;
    const bool horiz = (dir < 2);
    const bool fwd   = ((dir & 1) == 0);
    const float* Wk = (dir == 0) ? wk_hf : (dir == 1) ? wk_hb : (dir == 2) ? wk_vf : wk_vb;
    const float* Wr = (dir == 0) ? wr_hf : (dir == 1) ? wr_hb : (dir == 2) ? wr_vf : wr_vb;
    const int ch0 = dir * 256;

    const int tid  = threadIdx.x;
    const int lane = tid & 31;
    const int wave = tid >> 5;
    const int hi   = lane >> 4;
    const int lr   = lane & 15;
    const int kb   = hi * 8;

    // ---- preload bf16 B-fragments of Wk (64x256) and Wr (256x256) into VGPRs ----
    v16bf wkf[2][2];   // [ntile][ktile]
    v16bf wrf[2][8];   // [ntile][ktile]
    #pragma unroll
    for (int nt = 0; nt < 2; ++nt) {
        const int n = wave * 32 + nt * 16 + lr;
        #pragma unroll
        for (int kt = 0; kt < 2; ++kt) {
            v16bf f;
            #pragma unroll
            for (int j = 0; j < 8; ++j) {
                const int k = kt * 32 + kb + ((j < 4) ? 2 * j : 16 + 2 * (j - 4));
                f[2 * j]     = (__bf16)Wk[k * U_DIM + n];
                f[2 * j + 1] = (__bf16)Wk[(k + 1) * U_DIM + n];
            }
            wkf[nt][kt] = f;
        }
        #pragma unroll
        for (int kt = 0; kt < 8; ++kt) {
            v16bf f;
            #pragma unroll
            for (int j = 0; j < 8; ++j) {
                const int k = kt * 32 + kb + ((j < 4) ? 2 * j : 16 + 2 * (j - 4));
                f[2 * j]     = (__bf16)Wr[k * U_DIM + n];
                f[2 * j + 1] = (__bf16)Wr[(k + 1) * U_DIM + n];
            }
            wrf[nt][kt] = f;
        }
    }

    const int pos0 = fwd ? 0 : 127;
    const int sgn  = fwd ? 1 : -1;

    // ---- x staging: thread -> (row m, 4 channels); running pointer, const delta ----
    const int sm = tid >> 4;
    const int sc = (tid & 15) * 4;
    const int ssidx = stile * 16 + sm;
    long xbase; int xstride;
    if (horiz) { xbase = (long)ssidx * 8192 + sc;                                      xstride = 64;   }
    else       { xbase = (long)(ssidx >> 7) * 1048576 + (long)(ssidx & 127) * 64 + sc; xstride = 8192; }
    const float* xptr = x + xbase + (long)pos0 * xstride;
    const long   xdelta = (long)sgn * xstride;

    // ---- output: running byte offsets, constant per-step delta ----
    const int ostride = horiz ? 1024 : 131072;
    const int odelta  = sgn * ostride * 4;          // bytes per step
    unsigned ooff[2][8];
    #pragma unroll
    for (int nt = 0; nt < 2; ++nt)
        #pragma unroll
        for (int v = 0; v < 8; ++v) {
            const int m    = v + hi * 8;
            const int sidx = stile * 16 + m;
            const int ch   = ch0 + wave * 32 + nt * 16 + lr;
            const int base = horiz ? sidx * 131072 + ch
                                   : (sidx >> 7) * 16777216 + (sidx & 127) * 1024 + ch;
            ooff[nt][v] = (unsigned)(base + pos0 * ostride) * 4u;
        }

    // ---- prologue: h_{-1} = 0, stage x_0 ----
    for (int i = tid; i < NT_SEQ * HROW; i += 256)
        ((__bf16*)hlds[0])[i] = (__bf16)0.0f;
    {
        const float4 xv = *(const float4*)xptr;
        __bf16* dst = &xlds[0][sm][sc];
        dst[0] = (__bf16)xv.x; dst[1] = (__bf16)xv.y;
        dst[2] = (__bf16)xv.z; dst[3] = (__bf16)xv.w;
    }
    float hreg[2][8];
    #pragma unroll
    for (int nt = 0; nt < 2; ++nt)
        #pragma unroll
        for (int v = 0; v < 8; ++v) hreg[nt][v] = 0.0f;
    __syncthreads();

    // ---- recurrence ----
    for (int t = 0; t < T_LEN; ++t) {
        const int cur = t & 1, nxt = cur ^ 1;

        v8f acc0 = {}, acc1 = {};
        // input projection: (16x64) x (64x32-per-wave)
        #pragma unroll
        for (int kt = 0; kt < 2; ++kt) {
            v16bf a = load_frag16(&xlds[cur][lr][0], kt * 32 + kb);
            acc0 = __builtin_amdgcn_wmma_f32_16x16x32_bf16(false, a, false, wkf[0][kt], (short)0, acc0, false, false);
            acc1 = __builtin_amdgcn_wmma_f32_16x16x32_bf16(false, a, false, wkf[1][kt], (short)0, acc1, false, false);
        }
        // recurrent matmul: (16x256) x (256x32-per-wave)
        #pragma unroll
        for (int kt = 0; kt < 8; ++kt) {
            v16bf a = load_frag16(&hlds[cur][lr][0], kt * 32 + kb);
            acc0 = __builtin_amdgcn_wmma_f32_16x16x32_bf16(false, a, false, wrf[0][kt], (short)0, acc0, false, false);
            acc1 = __builtin_amdgcn_wmma_f32_16x16x32_bf16(false, a, false, wrf[1][kt], (short)0, acc1, false, false);
        }

        // leaky-integrator update + streaming (non-temporal) global store (f32)
        #pragma unroll
        for (int nt = 0; nt < 2; ++nt)
            #pragma unroll
            for (int v = 0; v < 8; ++v) {
                const float a = (nt == 0) ? acc0[v] : acc1[v];
                const float h = fmaf(0.1f, hreg[nt][v], 0.9f * fast_tanh(a));
                hreg[nt][v] = h;
                __builtin_nontemporal_store(h, (float*)((char*)out + ooff[nt][v]));
                ooff[nt][v] += odelta;
            }

        // publish h_t (bf16) into the next LDS buffer for step t+1's A-matrix
        {
            const int n0 = wave * 32 + lr;
            #pragma unroll
            for (int nt = 0; nt < 2; ++nt)
                #pragma unroll
                for (int v = 0; v < 8; ++v)
                    hlds[nxt][v + hi * 8][n0 + nt * 16] = (__bf16)hreg[nt][v];
        }

        // stage x_{t+1}; prefetch x_{t+2}
        if (t < T_LEN - 1) {
            const float4 xv = *(const float4*)(xptr + xdelta);
            __bf16* dst = &xlds[nxt][sm][sc];
            dst[0] = (__bf16)xv.x; dst[1] = (__bf16)xv.y;
            dst[2] = (__bf16)xv.z; dst[3] = (__bf16)xv.w;
        }
        if (t < T_LEN - 2)
            __builtin_prefetch(xptr + 2 * xdelta, 0, 1);
        xptr += xdelta;
        __syncthreads();
    }
}

extern "C" void kernel_launch(void* const* d_in, const int* in_sizes, int n_in,
                              void* d_out, int out_size, void* d_ws, size_t ws_size,
                              hipStream_t stream) {
    (void)in_sizes; (void)n_in; (void)out_size; (void)d_ws; (void)ws_size;
    const float* x     = (const float*)d_in[0];
    const float* wk_hf = (const float*)d_in[1];
    const float* wr_hf = (const float*)d_in[2];
    const float* wk_hb = (const float*)d_in[3];
    const float* wr_hb = (const float*)d_in[4];
    const float* wk_vf = (const float*)d_in[5];
    const float* wr_vf = (const float*)d_in[6];
    const float* wk_vb = (const float*)d_in[7];
    const float* wr_vb = (const float*)d_in[8];
    float* out = (float*)d_out;

    // 4 directions x 64 sequence-tiles; 256 threads = 8 wave32s per WG
    biesn2d_kernel<<<dim3(256), dim3(256), 0, stream>>>(
        x, wk_hf, wr_hf, wk_hb, wr_hb, wk_vf, wr_vf, wk_vb, wr_vb, out);
}
